// LogicGatedSNN_67808943669909
// MI455X (gfx1250) — matrix operand order repaired and checked by reading, not compiled, not run on hardware
//
#include <hip/hip_runtime.h>
#include <hip/hip_bf16.h>

typedef __attribute__((ext_vector_type(16))) _Float16 v16h;
typedef __attribute__((ext_vector_type(8)))  float    v8f;
typedef __attribute__((ext_vector_type(4)))  float    f4v;

#define IN_F   8192
#define OUT_F  8192
#define KSPLIT 8
#define KCHUNK (IN_F / KSPLIT)   // 1024 K per wave

// ---------------------------------------------------------------------------
// Kernel 0: zero the scratch (current[OUT_F] + flag[1]); ws is poisoned 0xAA
// by the harness, and atomicAdd accumulation requires zeros each call.
// ---------------------------------------------------------------------------
__global__ __launch_bounds__(256) void snn_init_ws(float* __restrict__ ws) {
    int i = blockIdx.x * 256 + threadIdx.x;
    if (i < OUT_F + 1) ws[i] = 0.0f;
}

// ---------------------------------------------------------------------------
// Kernel 1: binarize + GEMV via WMMA.
// One wave handles a 16-output-row tile over a 1024-wide K chunk.
//   A (16x32 f16)  = spike slice broadcast to all 16 rows
//   B (32x16 f16)  = (states > thr) transposed tile, column n = output row
//   C[m][n] = current[rowbase+n]  (independent of m) -> read acc VGPR0.
// Exact: operands are 0/1 (f16-exact), accumulation in f32.
// ---------------------------------------------------------------------------
__global__ __launch_bounds__(256) void snn_gemv_wmma(
    const float* __restrict__ spike,
    const float* __restrict__ states,
    const int*   __restrict__ thr_p,
    float*       __restrict__ current) {
    const int lane    = threadIdx.x & 31;
    const int wave    = (int)((blockIdx.x * blockDim.x + threadIdx.x) >> 5);
    const int tile    = wave / KSPLIT;          // 0..511 row tiles
    const int kseg    = wave % KSPLIT;          // 0..7  K segments
    const int rowbase = tile * 16;
    const int n       = lane & 15;              // output row within tile
    const int hi      = lane >> 4;              // lane half select
    const float thr   = (float)thr_p[0];

    const float* wrow = states + (size_t)(rowbase + n) * IN_F;

    v8f acc = {};
    const int k0 = kseg * KCHUNK;
    for (int kb = k0; kb < k0 + KCHUNK; kb += 32) {
        // ---- A fragment: spike broadcast (layout: lanes<16 -> K 0-15,
        //      lanes>=16 -> K 16-31 relative to kb)
        v16h a;
        {
            const f4v* sp0 = (const f4v*)(spike + kb + hi * 16);
            const f4v* sp1 = (const f4v*)(spike + kb + hi * 16 + 8);
            f4v s0 = sp0[0], s1 = sp0[1];
            f4v s2 = sp1[0], s3 = sp1[1];
#pragma unroll
            for (int e = 0; e < 4; ++e) {
                a[e]      = (_Float16)s0[e];
                a[4 + e]  = (_Float16)s1[e];
                a[8 + e]  = (_Float16)s2[e];
                a[12 + e] = (_Float16)s3[e];
            }
        }
        // ---- B fragment: binarized W^T (lanes<16 -> K 0-15, lanes>=16 -> K 16-31)
        v16h b;
        {
            const f4v* wp = (const f4v*)(wrow + kb + hi * 16);
#pragma unroll
            for (int q = 0; q < 4; ++q) {
                f4v w4 = wp[q];
#pragma unroll
                for (int e = 0; e < 4; ++e)
                    b[q * 4 + e] = (w4[e] > thr) ? (_Float16)1.0f : (_Float16)0.0f;
            }
        }
        acc = __builtin_amdgcn_wmma_f32_16x16x32_f16(
            /*neg_a=*/false, a, /*neg_b=*/false, b,
            /*c_mod=*/(short)0, acc, /*reuse_a=*/false, /*reuse_b=*/false);
    }
    // acc VGPR0: lane L (<16) holds current[rowbase + L] (lanes 16-31 duplicate).
    if (lane < 16) atomicAdd(&current[rowbase + lane], acc[0]);
}

// ---------------------------------------------------------------------------
// Kernel 2: flag = (max(current) < 0.1). current >= 0 so init with 0.
// ---------------------------------------------------------------------------
__global__ __launch_bounds__(256) void snn_maxflag(const float* __restrict__ current,
                                                   float* __restrict__ flag) {
    __shared__ float red[256];
    float m = 0.0f;
    for (int i = threadIdx.x; i < OUT_F; i += 256) m = fmaxf(m, current[i]);
    red[threadIdx.x] = m;
    __syncthreads();
    for (int s = 128; s > 0; s >>= 1) {
        if ((int)threadIdx.x < s)
            red[threadIdx.x] = fmaxf(red[threadIdx.x], red[threadIdx.x + s]);
        __syncthreads();
    }
    if (threadIdx.x == 0) flag[0] = (red[0] < 0.1f) ? 1.0f : 0.0f;
}

// ---------------------------------------------------------------------------
// Kernel 3: fused per-row LIF update + streaming trace update.
// One 256-thread block per output row; float4 (b128) loads/stores.
// ---------------------------------------------------------------------------
__global__ __launch_bounds__(256) void snn_update(
    const float* __restrict__ spike_in,
    const float* __restrict__ trace_in,
    const float* __restrict__ membrane,
    const float* __restrict__ athr,
    const float* __restrict__ noise,
    const float* __restrict__ current,
    const float* __restrict__ flag,
    float*       __restrict__ out) {
    const int row = blockIdx.x;

    float cur = current[row];
    if (flag[0] > 0.5f) cur += fabsf(noise[row]) * 0.5f;
    const float v   = membrane[row] * 0.8f + cur;
    const float at  = athr[row];
    const float spk = (v >= at) ? 1.0f : 0.0f;

    float* out_spikes = out;
    float* out_v      = out + OUT_F;
    float* out_trace  = out + 2 * OUT_F;
    float* out_thr    = out_trace + (size_t)OUT_F * IN_F;

    if (threadIdx.x == 0) {
        out_spikes[row] = spk;
        out_v[row]      = v * (1.0f - spk) * 0.2f;
        out_thr[row]    = fminf(fmaxf(at + (spk - 0.1f) * 0.1f, 0.1f), 10.0f);
    }

    const f4v* tr = (const f4v*)(trace_in + (size_t)row * IN_F);
    const f4v* si = (const f4v*)spike_in;
    f4v*       to = (f4v*)(out_trace + (size_t)row * IN_F);
    for (int i = threadIdx.x; i < IN_F / 4; i += 256) {
        f4v t = tr[i];
        f4v s = si[i];
        f4v r;
#pragma unroll
        for (int e = 0; e < 4; ++e)
            r[e] = fminf(fmaxf(t[e] * 0.85f + spk * s[e], 0.0f), 5.0f);
        to[i] = r;
    }
}

// ---------------------------------------------------------------------------
extern "C" void kernel_launch(void* const* d_in, const int* in_sizes, int n_in,
                              void* d_out, int out_size, void* d_ws, size_t ws_size,
                              hipStream_t stream) {
    const float* spike_input = (const float*)d_in[0];   // [8192]
    const float* synapse     = (const float*)d_in[1];   // [8192*8192]
    const float* membrane    = (const float*)d_in[2];   // [8192]
    const float* athr        = (const float*)d_in[3];   // [8192]
    const float* elig        = (const float*)d_in[4];   // [8192*8192]
    const float* noise       = (const float*)d_in[5];   // [8192]
    const int*   thr_p       = (const int*)d_in[6];     // scalar 50

    float* ws_current = (float*)d_ws;                   // [8192] partial sums
    float* ws_flag    = ws_current + OUT_F;             // [1] noise flag
    float* out        = (float*)d_out;

    // 0) zero scratch accumulators
    snn_init_ws<<<(OUT_F + 1 + 255) / 256, 256, 0, stream>>>(ws_current);

    // 1) WMMA GEMV: (OUT_F/16) row tiles * KSPLIT K-segments waves
    {
        const int waves  = (OUT_F / 16) * KSPLIT;       // 4096
        const int blocks = waves * 32 / 256;            // 512
        snn_gemv_wmma<<<blocks, 256, 0, stream>>>(spike_input, synapse, thr_p,
                                                  ws_current);
    }

    // 2) global max -> noise flag
    snn_maxflag<<<1, 256, 0, stream>>>(ws_current, ws_flag);

    // 3) fused LIF + eligibility-trace stream (dominant 512 MB of traffic)
    snn_update<<<OUT_F, 256, 0, stream>>>(spike_input, elig, membrane, athr,
                                          noise, ws_current, ws_flag, out);
}
